// ImplicitNegativeLoss_13314398617875
// MI455X (gfx1250) — compile-verified, matrix operand order checked by reading, not run on hardware
//
#include <hip/hip_runtime.h>

// ---------------------------------------------------------------------------
// In-batch InfoNCE loss for MI455X (gfx1250, wave32, WMMA).
//   u,p: [8192,256] fp32 -> normalize -> sim = (u_n @ p_n^T) * 5
//   loss = mean_i( log(sum_j exp(sim_ij)) - sim_ii )
// GEMM via v_wmma_f32_16x16x32_bf16 with bf16 hi/lo split
// (sim ~= hi*hi + hi*lo + lo*hi) for ~fp32 accuracy at tensor-core rate.
// Two i-tiles per workgroup: each B fragment feeds 6 WMMAs (2 indep. chains),
// halving L2 traffic vs one-tile blocking and hiding WMMA RAW latency.
// ---------------------------------------------------------------------------

typedef __attribute__((ext_vector_type(16))) __bf16 v16bf;
typedef __attribute__((ext_vector_type(8)))  float  v8f;

#define BATCH    8192
#define DIM      256
#define NTILES   (BATCH / 16)    // 512 column tiles of 16
#define NBLOCKS  (NTILES / 2)    // 256 workgroups, 2 row-tiles each
#define INV_TEMP 5.0f            // 1 / 0.2

union V16 { uint4 q[2]; v16bf v; };

static __device__ __forceinline__ unsigned short f32_to_bf16(float f) {
  unsigned int x = __float_as_uint(f);
  x += 0x7FFFu + ((x >> 16) & 1u);          // round-to-nearest-even
  return (unsigned short)(x >> 16);
}
static __device__ __forceinline__ float bf16_to_f32(unsigned short h) {
  return __uint_as_float(((unsigned int)h) << 16);
}

// ---------------------------------------------------------------------------
// Kernel 1: L2-normalize rows, split each value into bf16 hi + bf16 lo.
// One wave per row (8 rows per 256-thread block).
// ---------------------------------------------------------------------------
__global__ __launch_bounds__(256)
void normalize_split_kernel(const float* __restrict__ x,
                            unsigned short* __restrict__ hi,
                            unsigned short* __restrict__ lo) {
  const int wave = threadIdx.x >> 5;
  const int lane = threadIdx.x & 31;
  const int row  = blockIdx.x * 8 + wave;

  const float* xr = x + (long)row * DIM;
  float vals[8];
  float ss = 0.0f;
#pragma unroll
  for (int i = 0; i < 8; ++i) {
    vals[i] = xr[lane + 32 * i];
    ss += vals[i] * vals[i];
  }
#pragma unroll
  for (int m = 16; m >= 1; m >>= 1) ss += __shfl_xor(ss, m, 32);

  const float inv = 1.0f / fmaxf(sqrtf(ss), 1e-12f);

#pragma unroll
  for (int i = 0; i < 8; ++i) {
    const float v = vals[i] * inv;
    const unsigned short h = f32_to_bf16(v);
    const unsigned short l = f32_to_bf16(v - bf16_to_f32(h));
    hi[(long)row * DIM + lane + 32 * i] = h;
    lo[(long)row * DIM + lane + 32 * i] = l;
  }
}

// ---------------------------------------------------------------------------
// Kernel 2: per 32-row band (two 16-row WMMA tiles), GEMM over all 512
// column tiles with fused exp-sum + diagonal extraction.
// Block = 256 threads = 8 waves; wave w handles j-tiles w, w+8, ...
// Each B fragment is reused against both A tiles (acc0 / acc1).
// ---------------------------------------------------------------------------
__global__ __launch_bounds__(256)
void simlse_kernel(const unsigned short* __restrict__ uh,
                   const unsigned short* __restrict__ ul,
                   const unsigned short* __restrict__ ph,
                   const unsigned short* __restrict__ pl,
                   float* __restrict__ partials) {
  __shared__ unsigned short sA_hi[32 * DIM];   // 16 KB
  __shared__ unsigned short sA_lo[32 * DIM];   // 16 KB
  __shared__ float s_rowsum[8][32];
  __shared__ float s_diag[32];

  const int tid     = threadIdx.x;
  const int wave    = tid >> 5;
  const int lane    = tid & 31;
  const int i_tile0 = blockIdx.x * 2;
  const int i_tile1 = i_tile0 + 1;
  const int row0    = i_tile0 * 16;            // 32 consecutive rows staged

  // --- cooperative stage of both A bands (hi+lo) into LDS ----------------
  {
    const uint4* gh = (const uint4*)(uh + (long)row0 * DIM);
    const uint4* gl = (const uint4*)(ul + (long)row0 * DIM);
    uint4* lh = (uint4*)sA_hi;
    uint4* ll = (uint4*)sA_lo;
    for (int c = tid; c < 32 * DIM / 8; c += 256) {
      lh[c] = gh[c];
      ll[c] = gl[c];
    }
  }
  __syncthreads();

  // Per-lane ISA layout for 16-bit WMMA A/B operands:
  //   lanes 0-15 : row/col = lane,    K chunks {kb*32+0..7, kb*32+16..23}
  //   lanes 16-31: row/col = lane-16, K chunks {kb*32+8..15, kb*32+24..31}
  const int mr    = lane & 15;
  const int kbase = (lane >> 4) * 8;

  float es0[8], es1[8];
#pragma unroll
  for (int r = 0; r < 8; ++r) { es0[r] = 0.0f; es1[r] = 0.0f; }

  for (int j_tile = wave; j_tile < NTILES; j_tile += 8) {
    const long jrow = (long)(j_tile * 16 + mr) * DIM;
    const unsigned short* bh = ph + jrow;
    const unsigned short* bl = pl + jrow;

    v8f a0 = {};
    v8f a1 = {};
#pragma unroll
    for (int kb = 0; kb < 8; ++kb) {
      const int k0 = kb * 32 + kbase;
      V16 Bh, Bl, A0h, A0l, A1h, A1l;
      Bh.q[0]  = *(const uint4*)(bh + k0);
      Bh.q[1]  = *(const uint4*)(bh + k0 + 16);
      Bl.q[0]  = *(const uint4*)(bl + k0);
      Bl.q[1]  = *(const uint4*)(bl + k0 + 16);
      A0h.q[0] = *(const uint4*)(sA_hi + mr * DIM + k0);
      A0h.q[1] = *(const uint4*)(sA_hi + mr * DIM + k0 + 16);
      A0l.q[0] = *(const uint4*)(sA_lo + mr * DIM + k0);
      A0l.q[1] = *(const uint4*)(sA_lo + mr * DIM + k0 + 16);
      A1h.q[0] = *(const uint4*)(sA_hi + (16 + mr) * DIM + k0);
      A1h.q[1] = *(const uint4*)(sA_hi + (16 + mr) * DIM + k0 + 16);
      A1l.q[0] = *(const uint4*)(sA_lo + (16 + mr) * DIM + k0);
      A1l.q[1] = *(const uint4*)(sA_lo + (16 + mr) * DIM + k0 + 16);

      // Two independent accumulation chains per B fragment.
      a0 = __builtin_amdgcn_wmma_f32_16x16x32_bf16(
          false, A0h.v, false, Bh.v, (short)0, a0, false, false);
      a1 = __builtin_amdgcn_wmma_f32_16x16x32_bf16(
          false, A1h.v, false, Bh.v, (short)0, a1, false, false);
      a0 = __builtin_amdgcn_wmma_f32_16x16x32_bf16(
          false, A0h.v, false, Bl.v, (short)0, a0, false, false);
      a1 = __builtin_amdgcn_wmma_f32_16x16x32_bf16(
          false, A1h.v, false, Bl.v, (short)0, a1, false, false);
      a0 = __builtin_amdgcn_wmma_f32_16x16x32_bf16(
          false, A0l.v, false, Bh.v, (short)0, a0, false, false);
      a1 = __builtin_amdgcn_wmma_f32_16x16x32_bf16(
          false, A1l.v, false, Bh.v, (short)0, a1, false, false);
    }

    // Diagonal tiles: sim_ii sits at (VGPR r, lane r) for rows 0-7 and
    // (VGPR r, lane r+24) for rows 8-15 of the C/D layout.
    if (j_tile == i_tile0) {
#pragma unroll
      for (int r = 0; r < 8; ++r) {
        const float v = a0[r] * INV_TEMP;
        if (lane == r)      s_diag[r]     = v;
        if (lane == r + 24) s_diag[r + 8] = v;
      }
    }
    if (j_tile == i_tile1) {
#pragma unroll
      for (int r = 0; r < 8; ++r) {
        const float v = a1[r] * INV_TEMP;
        if (lane == r)      s_diag[16 + r]     = v;
        if (lane == r + 24) s_diag[16 + r + 8] = v;
      }
    }

    // |sim/T| <= 5 -> exp never overflows; skip max subtraction.
#pragma unroll
    for (int r = 0; r < 8; ++r) {
      es0[r] += __expf(a0[r] * INV_TEMP);
      es1[r] += __expf(a1[r] * INV_TEMP);
    }
  }

  // --- reduce exp-sums: lanes 0-15 hold row r, lanes 16-31 hold row r+8 ---
#pragma unroll
  for (int r = 0; r < 8; ++r) {
    float s = es0[r];
    s += __shfl_xor(s, 1, 32);
    s += __shfl_xor(s, 2, 32);
    s += __shfl_xor(s, 4, 32);
    s += __shfl_xor(s, 8, 32);
    if (lane == 0)  s_rowsum[wave][r]     = s;
    if (lane == 16) s_rowsum[wave][r + 8] = s;

    float t = es1[r];
    t += __shfl_xor(t, 1, 32);
    t += __shfl_xor(t, 2, 32);
    t += __shfl_xor(t, 4, 32);
    t += __shfl_xor(t, 8, 32);
    if (lane == 0)  s_rowsum[wave][16 + r]     = t;
    if (lane == 16) s_rowsum[wave][16 + r + 8] = t;
  }
  __syncthreads();

  // First wave: 32 rows -> per-block partial loss.
  if (tid < 32) {
    float tot = 0.0f;
#pragma unroll
    for (int w = 0; w < 8; ++w) tot += s_rowsum[w][tid];
    float loss = __logf(tot) - s_diag[tid];   // lse_i - pos_sim_i
    loss += __shfl_xor(loss, 1, 32);
    loss += __shfl_xor(loss, 2, 32);
    loss += __shfl_xor(loss, 4, 32);
    loss += __shfl_xor(loss, 8, 32);
    loss += __shfl_xor(loss, 16, 32);
    if (tid == 0) partials[blockIdx.x] = loss;
  }
}

// ---------------------------------------------------------------------------
// Kernel 3: deterministic tree reduction of the 256 per-band partials.
// ---------------------------------------------------------------------------
__global__ __launch_bounds__(256)
void reduce_kernel(const float* __restrict__ partials, float* __restrict__ out) {
  __shared__ float s[256];
  const int tid = threadIdx.x;
  float v = 0.0f;
  for (int i = tid; i < NBLOCKS; i += 256) v += partials[i];
  s[tid] = v;
  __syncthreads();
  for (int off = 128; off > 0; off >>= 1) {
    if (tid < off) s[tid] += s[tid + off];
    __syncthreads();
  }
  if (tid == 0) out[0] = s[0] / (float)BATCH;
}

// ---------------------------------------------------------------------------
extern "C" void kernel_launch(void* const* d_in, const int* in_sizes, int n_in,
                              void* d_out, int out_size, void* d_ws, size_t ws_size,
                              hipStream_t stream) {
  const float* u = (const float*)d_in[0];
  const float* p = (const float*)d_in[1];
  float* out = (float*)d_out;

  const long N = (long)BATCH * DIM;
  unsigned short* uh = (unsigned short*)d_ws;
  unsigned short* ul = uh + N;
  unsigned short* ph = ul + N;
  unsigned short* pl = ph + N;
  float* partials = (float*)(pl + N);

  normalize_split_kernel<<<BATCH / 8, 256, 0, stream>>>(u, uh, ul);
  normalize_split_kernel<<<BATCH / 8, 256, 0, stream>>>(p, ph, pl);
  simlse_kernel<<<NBLOCKS, 256, 0, stream>>>(uh, ul, ph, pl, partials);
  reduce_kernel<<<1, 256, 0, stream>>>(partials, out);
}
